// LocalGlobal_16243566313718
// MI455X (gfx1250) — matrix operand order repaired and checked by reading, not compiled, not run on hardware
//
#include <hip/hip_runtime.h>
#include <hip/hip_bf16.h>
#include <stdint.h>

// ---------------- types / helpers ----------------
typedef __attribute__((ext_vector_type(16))) __bf16 v16bf_t;
typedef __attribute__((ext_vector_type(8)))  float  v8f_t;
typedef unsigned short u16_t;

__device__ __forceinline__ u16_t f2bf(float f) {
  union { float f; uint32_t u; } v; v.f = f;
  uint32_t u = v.u;
  return (u16_t)((u + 0x7FFFu + ((u >> 16) & 1u)) >> 16);  // RNE
}

union FragBF {
  v16bf_t bf;
  uint4 q[2];
};

// ---------------- small batched GEMM: 64x64 tile, 8 waves, 16x32 per wave ----------------
// Used for head-batched attention GEMMs (M=64 or N=64 per batch).
// A: (M,K) bf16 row-major, W: (N,K) bf16 row-major. C = act(scale*A@W^T + bias) (+resid)
__global__ __launch_bounds__(256) void gemm_bf16_kernel(
    const u16_t* __restrict__ A, long lda, long sAb, long sAh,
    const u16_t* __restrict__ Bw, long ldb, long sBb, long sBh,
    const float* __restrict__ bias,
    const float* __restrict__ resid,
    float* __restrict__ Cf, u16_t* __restrict__ Ch,
    long ldc, long sCb, long sCh,
    int Kd, int Hsplit, float scale, int relu)
{
  __shared__ __attribute__((aligned(16))) u16_t As[64 * 40];
  __shared__ __attribute__((aligned(16))) u16_t Bs[64 * 40];

  const int z = blockIdx.z;
  const int bb = z / Hsplit, hh = z % Hsplit;
  A  += (long)bb * sAb + (long)hh * sAh;
  Bw += (long)bb * sBb + (long)hh * sBh;
  const long cbase = (long)bb * sCb + (long)hh * sCh;

  const int tid = threadIdx.x;
  const int lane = tid & 31, wave = tid >> 5;
  const int wm = wave & 3, wn = wave >> 2;        // 4 x 2 wave grid
  const long rowg = (long)blockIdx.x * 64;
  const long colg = (long)blockIdx.y * 64;
  const int lr = tid >> 2;                        // 0..63 (tile row)
  const int lc = (tid & 3) * 8;                   // 0,8,16,24 (k chunk)

  v8f_t acc0 = {}; v8f_t acc1 = {};

  for (int k0 = 0; k0 < Kd; k0 += 32) {
    uint4 av = *(const uint4*)(A  + (rowg + lr) * lda + k0 + lc);
    uint4 bv = *(const uint4*)(Bw + (colg + lr) * ldb + k0 + lc);
    *(uint4*)&As[lr * 40 + lc] = av;
    *(uint4*)&Bs[lr * 40 + lc] = bv;
    if (k0 + 32 < Kd) {
      __builtin_prefetch(A  + (rowg + lr) * lda + k0 + 32 + lc, 0, 0);
      __builtin_prefetch(Bw + (colg + lr) * ldb + k0 + 32 + lc, 0, 0);
    }
    __syncthreads();

    FragBF a, b0, b1;
    const int arow = wm * 16 + (lane & 15);
    const int ka = (lane < 16) ? 0 : 8;           // A: lanes0-15 K{0-7,16-23}, lanes16-31 +8
    a.q[0] = *(const uint4*)&As[arow * 40 + ka];
    a.q[1] = *(const uint4*)&As[arow * 40 + 16 + ka];
    const int kb = (lane < 16) ? 0 : 8;           // B: lanes0-15 K{0-7,16-23}, lanes16-31 +8
    const int n0 = wn * 32 + (lane & 15);
    b0.q[0] = *(const uint4*)&Bs[n0 * 40 + kb];
    b0.q[1] = *(const uint4*)&Bs[n0 * 40 + kb + 16];
    const int n1 = n0 + 16;
    b1.q[0] = *(const uint4*)&Bs[n1 * 40 + kb];
    b1.q[1] = *(const uint4*)&Bs[n1 * 40 + kb + 16];

    acc0 = __builtin_amdgcn_wmma_f32_16x16x32_bf16(false, a.bf, false, b0.bf, (short)0, acc0, false, false);
    acc1 = __builtin_amdgcn_wmma_f32_16x16x32_bf16(false, a.bf, false, b1.bf, (short)0, acc1, false, false);
    __syncthreads();
  }

  const int rofs = (lane < 16) ? 0 : 8;           // C: vgpr r -> row r / r+8
  const int cl = lane & 15;
  for (int sub = 0; sub < 2; ++sub) {
    v8f_t acc = sub ? acc1 : acc0;
    const long col = colg + wn * 32 + sub * 16 + cl;
    for (int r = 0; r < 8; ++r) {
      const long row = rowg + wm * 16 + r + rofs;
      float v = acc[r] * scale;
      if (bias)  v += bias[col];
      if (relu)  v = v > 0.f ? v : 0.f;
      if (resid) v += resid[row * ldc + col];
      const long ci = cbase + row * ldc + col;
      if (Cf) Cf[ci] = v;
      if (Ch) Ch[ci] = f2bf(v);
    }
  }
}

// ---------------- big GEMM: 128x128 tile, 8 waves (4x2), 32x64 per wave ----------------
// 8 WMMAs per 12 ds_load_b128 per K-step (vs 2:6 in the small variant).
__global__ __launch_bounds__(256) void gemm_bf16_big_kernel(
    const u16_t* __restrict__ A, long lda,
    const u16_t* __restrict__ Bw, long ldb,
    const float* __restrict__ bias,
    const float* __restrict__ resid,
    float* __restrict__ Cf, u16_t* __restrict__ Ch, long ldc,
    int Kd, float scale, int relu)
{
  __shared__ __attribute__((aligned(16))) u16_t As[128 * 40];
  __shared__ __attribute__((aligned(16))) u16_t Bs[128 * 40];

  const int tid = threadIdx.x;
  const int lane = tid & 31, wave = tid >> 5;
  const int wm = wave >> 1;                       // 0..3 -> row base wm*32
  const int wn = wave & 1;                        // 0..1 -> col base wn*64
  const long rowg = (long)blockIdx.x * 128;
  const long colg = (long)blockIdx.y * 128;

  // cooperative tile load mapping: 128x32 elems = 2 x b128 per thread
  const int r0 = tid >> 1;                        // 0..127
  const int c0 = (tid & 1) * 8;                   // 0 or 8
  const int c1 = c0 + 16;                         // 16 or 24

  v8f_t acc[2][4] = {};

  for (int k0 = 0; k0 < Kd; k0 += 32) {
    uint4 av0 = *(const uint4*)(A  + (rowg + r0) * lda + k0 + c0);
    uint4 av1 = *(const uint4*)(A  + (rowg + r0) * lda + k0 + c1);
    uint4 bv0 = *(const uint4*)(Bw + (colg + r0) * ldb + k0 + c0);
    uint4 bv1 = *(const uint4*)(Bw + (colg + r0) * ldb + k0 + c1);
    *(uint4*)&As[r0 * 40 + c0] = av0;
    *(uint4*)&As[r0 * 40 + c1] = av1;
    *(uint4*)&Bs[r0 * 40 + c0] = bv0;
    *(uint4*)&Bs[r0 * 40 + c1] = bv1;
    if (k0 + 32 < Kd) {                           // cover next tile's latency (global_prefetch_b8)
      __builtin_prefetch(A  + (rowg + r0) * lda + k0 + 32 + c0, 0, 0);
      __builtin_prefetch(Bw + (colg + r0) * ldb + k0 + 32 + c0, 0, 0);
    }
    __syncthreads();

    FragBF a[2], bf[4];
    const int ka = (lane < 16) ? 0 : 8;
    const int kb = (lane < 16) ? 0 : 8;
    for (int i = 0; i < 2; ++i) {
      const int arow = wm * 32 + i * 16 + (lane & 15);
      a[i].q[0] = *(const uint4*)&As[arow * 40 + ka];
      a[i].q[1] = *(const uint4*)&As[arow * 40 + 16 + ka];
    }
    for (int j = 0; j < 4; ++j) {
      const int n = wn * 64 + j * 16 + (lane & 15);
      bf[j].q[0] = *(const uint4*)&Bs[n * 40 + kb];
      bf[j].q[1] = *(const uint4*)&Bs[n * 40 + kb + 16];
    }
    for (int i = 0; i < 2; ++i)
      for (int j = 0; j < 4; ++j)
        acc[i][j] = __builtin_amdgcn_wmma_f32_16x16x32_bf16(false, a[i].bf, false, bf[j].bf,
                                                            (short)0, acc[i][j], false, false);
    __syncthreads();
  }

  const int rofs = (lane < 16) ? 0 : 8;
  const int cl = lane & 15;
  for (int i = 0; i < 2; ++i) {
    for (int j = 0; j < 4; ++j) {
      const long col = colg + wn * 64 + j * 16 + cl;
      for (int r = 0; r < 8; ++r) {
        const long row = rowg + wm * 32 + i * 16 + r + rofs;
        float v = acc[i][j][r] * scale;
        if (bias)  v += bias[col];
        if (relu)  v = v > 0.f ? v : 0.f;
        if (resid) v += resid[row * ldc + col];
        const long ci = row * ldc + col;
        if (Cf) Cf[ci] = v;
        if (Ch) Ch[ci] = f2bf(v);
      }
    }
  }
}

// ---------------- KNN: top-16 nearest of 4096 points per (b,kpt) ----------------
__global__ __launch_bounds__(256) void knn_kernel(const float* __restrict__ kpt3d,
                                                  const float* __restrict__ pts,
                                                  int* __restrict__ idxout) {
  __shared__ float d[4096];
  __shared__ float rv[256];
  __shared__ int ri[256];
  const int blk = blockIdx.x;               // b*64 + kp
  const int b = blk >> 6;
  const float* P = pts + (long)b * 4096 * 3;
  const float qx = kpt3d[blk * 3 + 0], qy = kpt3d[blk * 3 + 1], qz = kpt3d[blk * 3 + 2];
  const int tid = threadIdx.x;
  for (int j = tid; j < 4096; j += 256) {
    float dx = P[j * 3 + 0] - qx, dy = P[j * 3 + 1] - qy, dz = P[j * 3 + 2] - qz;
    d[j] = dx * dx + dy * dy + dz * dz;
  }
  __syncthreads();
  for (int s = 0; s < 16; ++s) {
    float bv = 3.4e38f; int bi = 0x7fffffff;
    for (int j = tid; j < 4096; j += 256) {
      float v = d[j];
      if (v < bv) { bv = v; bi = j; }
    }
    rv[tid] = bv; ri[tid] = bi; __syncthreads();
    for (int o = 128; o; o >>= 1) {
      if (tid < o) {
        float v2 = rv[tid + o]; int i2 = ri[tid + o];
        if (v2 < rv[tid] || (v2 == rv[tid] && i2 < ri[tid])) { rv[tid] = v2; ri[tid] = i2; }
      }
      __syncthreads();
    }
    if (tid == 0) { idxout[blk * 16 + s] = ri[0]; d[ri[0]] = 3.4e38f; }
    __syncthreads();
  }
}

// gather neighbor features (bf16 -> cat_knn cols 0..255) and xyz (fp32)
__global__ __launch_bounds__(256) void gather_kernel(const int* __restrict__ idx,
                                                     const float* __restrict__ pts,
                                                     const float* __restrict__ objf,
                                                     float* __restrict__ knn_xyz,
                                                     u16_t* __restrict__ cat_knn) {
  const int blk = blockIdx.x;               // ((b*64+kp)*16 + k), 0..16383
  const int b = blk >> 10;
  const int id = idx[blk];
  const long src = (long)b * 4096 + id;
  const int c = threadIdx.x;
  cat_knn[(long)blk * 512 + c] = f2bf(objf[src * 256 + c]);
  if (c < 3) knn_xyz[blk * 3 + c] = pts[src * 3 + c];
}

// mlp3 layer 1: (rows,3) -> relu -> (rows,64) bf16
__global__ void mlp_l1_kernel(const float* __restrict__ X, const float* __restrict__ W,
                              const float* __restrict__ Bb, u16_t* __restrict__ Y, int rows) {
  const long gid = (long)blockIdx.x * 256 + threadIdx.x;
  if (gid >= (long)rows * 64) return;
  const long row = gid >> 6; const int o = gid & 63;
  float v = Bb[o] + W[o * 3] * X[row * 3] + W[o * 3 + 1] * X[row * 3 + 1] + W[o * 3 + 2] * X[row * 3 + 2];
  Y[gid] = f2bf(v > 0.f ? v : 0.f);
}

// softmax over rows of width W (fp32 in, bf16 out)
__global__ __launch_bounds__(256) void softmax_kernel(const float* __restrict__ S,
                                                      u16_t* __restrict__ P, int W) {
  __shared__ float red[256];
  const long row = blockIdx.x;
  const float* s = S + row * (long)W;
  u16_t* p = P + row * (long)W;
  const int tid = threadIdx.x;
  float m = -3.4e38f;
  for (int j = tid; j < W; j += 256) m = fmaxf(m, s[j]);
  red[tid] = m; __syncthreads();
  for (int o = 128; o; o >>= 1) { if (tid < o) red[tid] = fmaxf(red[tid], red[tid + o]); __syncthreads(); }
  m = red[0]; __syncthreads();
  float sum = 0.f;
  for (int j = tid; j < W; j += 256) sum += __expf(s[j] - m);
  red[tid] = sum; __syncthreads();
  for (int o = 128; o; o >>= 1) { if (tid < o) red[tid] += red[tid + o]; __syncthreads(); }
  const float inv = 1.f / red[0];
  for (int j = tid; j < W; j += 256) p[j] = f2bf(__expf(s[j] - m) * inv);
}

// LayerNorm over 256 cols, fp32 in -> bf16 out
__global__ __launch_bounds__(256) void ln_kernel(const float* __restrict__ X,
                                                 const float* __restrict__ g,
                                                 const float* __restrict__ b,
                                                 u16_t* __restrict__ Y) {
  __shared__ float r1[256], r2[256];
  const long row = blockIdx.x; const int c = threadIdx.x;
  const float x = X[row * 256 + c];
  r1[c] = x; r2[c] = x * x; __syncthreads();
  for (int o = 128; o; o >>= 1) { if (c < o) { r1[c] += r1[c + o]; r2[c] += r2[c + o]; } __syncthreads(); }
  const float m = r1[0] * (1.f / 256.f);
  const float var = r2[0] * (1.f / 256.f) - m * m;
  const float inv = rsqrtf(var + 1e-5f);
  Y[row * 256 + c] = f2bf((x - m) * inv * g[c] + b[c]);
}

// fp32 -> bf16 converters
__global__ void cvt_kernel(const float* __restrict__ s, u16_t* __restrict__ d, long n) {
  long i = (long)blockIdx.x * 256 + threadIdx.x;
  if (i < n) d[i] = f2bf(s[i]);
}
__global__ void cvt_strided_kernel(const float* __restrict__ s, long sld,
                                   u16_t* __restrict__ d, long dld, long rows, long cols) {
  long i = (long)blockIdx.x * 256 + threadIdx.x;
  if (i >= rows * cols) return;
  long r = i / cols, c = i % cols;
  d[r * dld + c] = f2bf(s[r * sld + c]);
}

// Weff[o][c] = sum_k fc_w1[o][k*256+c], bf16 out (query-collapse of the 4096-wide fc)
__global__ void weff_kernel(const float* __restrict__ w, u16_t* __restrict__ out) {
  const int i = blockIdx.x * 256 + threadIdx.x;   // 65536
  const int o = i >> 8, c = i & 255;
  float s = 0.f;
  for (int k = 0; k < 16; ++k) s += w[(long)o * 4096 + k * 256 + c];
  out[i] = f2bf(s);
}

// transpose V (cross-attn): KVh[(b*1024+t)*512 + 256 + d] -> Vt[b*262144 + d*1024 + t]
__global__ void vtrans_kernel(const u16_t* __restrict__ kvh, u16_t* __restrict__ vt) {
  long i = (long)blockIdx.x * 256 + threadIdx.x;  // 16*1024*256
  if (i >= (long)16 * 1024 * 256) return;
  long b = i >> 18, rem = i & ((1L << 18) - 1);
  long t = rem >> 8, dd = rem & 255;
  vt[b * 262144 + dd * 1024 + t] = kvh[(b * 1024 + t) * 512 + 256 + dd];
}
// transpose V (self-attn): qkv[(b*64+t)*768 + 512 + d] -> Vt2[b*16384 + d*64 + t]
__global__ void vtrans2_kernel(const u16_t* __restrict__ qkv, u16_t* __restrict__ vt2) {
  long i = (long)blockIdx.x * 256 + threadIdx.x;  // 16*64*256
  if (i >= (long)16 * 64 * 256) return;
  long b = i >> 14, rem = i & ((1L << 14) - 1);
  long t = rem >> 8, dd = rem & 255;
  vt2[b * 16384 + dd * 64 + t] = qkv[(b * 64 + t) * 768 + 512 + dd];
}

// g = mean over 64 tokens per (b,c)
__global__ __launch_bounds__(256) void meang_kernel(const float* __restrict__ x2, float* __restrict__ g) {
  const int b = blockIdx.x, c = threadIdx.x;
  float s = 0.f;
  for (int t = 0; t < 64; ++t) s += x2[((long)b * 64 + t) * 256 + c];
  g[b * 256 + c] = s * (1.f / 64.f);
}
// cat2 = [bf16(x2) | bf16(g broadcast)] (1024 x 512)
__global__ void cat2_kernel(const float* __restrict__ x2, const float* __restrict__ g,
                            u16_t* __restrict__ cat2) {
  long i = (long)blockIdx.x * 256 + threadIdx.x;  // 1024*512
  if (i >= (long)1024 * 512) return;
  long r = i >> 9; int c = (int)(i & 511);
  long b = r >> 6;
  float v = (c < 256) ? x2[r * 256 + c] : g[b * 256 + (c - 256)];
  cat2[i] = f2bf(v);
}

// ---------------- host ----------------
extern "C" void kernel_launch(void* const* d_in, const int* in_sizes, int n_in,
                              void* d_out, int out_size, void* d_ws, size_t ws_size,
                              hipStream_t stream) {
  (void)in_sizes; (void)n_in; (void)out_size; (void)ws_size;
  const float* kpt_feature = (const float*)d_in[0];
  const float* kpt_3d      = (const float*)d_in[1];
  const float* obj_feature = (const float*)d_in[2];
  const float* pts         = (const float*)d_in[3];
  // params (dict insertion order): d3[6], d4[6], d1[4], d2[4], blocks[0][26], blocks[1][26]
  const float* d3w1 = (const float*)d_in[4];  const float* d3b1 = (const float*)d_in[5];
  const float* d3w2 = (const float*)d_in[6];  const float* d3b2 = (const float*)d_in[7];
  const float* d3w3 = (const float*)d_in[8];  const float* d3b3 = (const float*)d_in[9];
  const float* d4w1 = (const float*)d_in[10]; const float* d4b1 = (const float*)d_in[11];
  const float* d4w2 = (const float*)d_in[12]; const float* d4b2 = (const float*)d_in[13];
  const float* d4w3 = (const float*)d_in[14]; const float* d4b3 = (const float*)d_in[15];
  const float* d1w1 = (const float*)d_in[16]; const float* d1b1 = (const float*)d_in[17];
  const float* d1w2 = (const float*)d_in[18]; const float* d1b2 = (const float*)d_in[19];
  const float* d2w1 = (const float*)d_in[20]; const float* d2b1 = (const float*)d_in[21];
  const float* d2w2 = (const float*)d_in[22]; const float* d2b2 = (const float*)d_in[23];
  const int BK1 = 50;  // only the LAST block's output matters (block 0 is dead code)
  const float* ca_win  = (const float*)d_in[BK1 + 0];  const float* ca_bin  = (const float*)d_in[BK1 + 1];
  const float* ca_wout = (const float*)d_in[BK1 + 2];  const float* ca_bout = (const float*)d_in[BK1 + 3];
  const float* sa_win  = (const float*)d_in[BK1 + 4];  const float* sa_bin  = (const float*)d_in[BK1 + 5];
  const float* sa_wout = (const float*)d_in[BK1 + 6];  const float* sa_bout = (const float*)d_in[BK1 + 7];
  const float* ln1_g = (const float*)d_in[BK1 + 8];  const float* ln1_b = (const float*)d_in[BK1 + 9];
  const float* ln2_g = (const float*)d_in[BK1 + 10]; const float* ln2_b = (const float*)d_in[BK1 + 11];
  const float* ln3_g = (const float*)d_in[BK1 + 12]; const float* ln3_b = (const float*)d_in[BK1 + 13];
  const float* ffn_w1 = (const float*)d_in[BK1 + 14]; const float* ffn_b1 = (const float*)d_in[BK1 + 15];
  const float* ffn_w2 = (const float*)d_in[BK1 + 16]; const float* ffn_b2 = (const float*)d_in[BK1 + 17];
  const float* fuse_w1 = (const float*)d_in[BK1 + 18]; const float* fuse_b1 = (const float*)d_in[BK1 + 19];
  const float* fuse_w2 = (const float*)d_in[BK1 + 20]; const float* fuse_b2 = (const float*)d_in[BK1 + 21];
  const float* fc_w1 = (const float*)d_in[BK1 + 22]; const float* fc_b1 = (const float*)d_in[BK1 + 23];
  const float* fc_w2 = (const float*)d_in[BK1 + 24]; const float* fc_b2 = (const float*)d_in[BK1 + 25];

  char* base = (char*)d_ws; size_t off = 0;
  auto alloc = [&](size_t bytes) -> void* {
    void* p = base + off; off += (bytes + 255) & ~(size_t)255; return p;
  };
  // bf16 weights
  u16_t* wb_d3w2 = (u16_t*)alloc(256 * 64 * 2);   u16_t* wb_d3w3 = (u16_t*)alloc(256 * 256 * 2);
  u16_t* wb_d4w2 = (u16_t*)alloc(256 * 64 * 2);   u16_t* wb_d4w3 = (u16_t*)alloc(256 * 256 * 2);
  u16_t* wb_d1w1 = (u16_t*)alloc(256 * 512 * 2);  u16_t* wb_d1w2 = (u16_t*)alloc(256 * 256 * 2);
  u16_t* wb_d2w1 = (u16_t*)alloc(256 * 512 * 2);  u16_t* wb_d2w2 = (u16_t*)alloc(256 * 256 * 2);
  u16_t* wb_cawin = (u16_t*)alloc(768 * 256 * 2); u16_t* wb_cawout = (u16_t*)alloc(256 * 256 * 2);
  u16_t* wb_sawin = (u16_t*)alloc(768 * 256 * 2); u16_t* wb_sawout = (u16_t*)alloc(256 * 256 * 2);
  u16_t* wb_ffn1 = (u16_t*)alloc(256 * 256 * 2);  u16_t* wb_ffn2 = (u16_t*)alloc(256 * 256 * 2);
  u16_t* wb_fuse1 = (u16_t*)alloc(256 * 512 * 2); u16_t* wb_fuse2 = (u16_t*)alloc(256 * 256 * 2);
  u16_t* wb_weff = (u16_t*)alloc(256 * 256 * 2);  u16_t* wb_fcw2 = (u16_t*)alloc(256 * 256 * 2);
  // activations
  int* idxb = (int*)alloc(16384 * 4);
  float* knn_xyz = (float*)alloc((size_t)16384 * 3 * 4);
  u16_t* kh1 = (u16_t*)alloc((size_t)1024 * 64 * 2);
  u16_t* kh2 = (u16_t*)alloc((size_t)1024 * 256 * 2);
  u16_t* cat_kpt = (u16_t*)alloc((size_t)1024 * 512 * 2);
  u16_t* t1 = (u16_t*)alloc((size_t)1024 * 256 * 2);
  float* kpt_comb = (float*)alloc((size_t)1024 * 256 * 4);
  u16_t* q1 = (u16_t*)alloc((size_t)1024 * 256 * 2);
  u16_t* Qh = (u16_t*)alloc((size_t)1024 * 256 * 2);
  u16_t* cat_knn = (u16_t*)alloc((size_t)16384 * 512 * 2);   // aliased by KVh later
  u16_t* nh1 = (u16_t*)alloc((size_t)16384 * 64 * 2);
  u16_t* nh2 = (u16_t*)alloc((size_t)16384 * 256 * 2);       // aliased by t2, Vt
  u16_t* kv  = (u16_t*)alloc((size_t)16384 * 256 * 2);       // aliased by P
  float* Sc  = (float*)alloc((size_t)16 * 4 * 64 * 1024 * 4);
  u16_t* attn_o = (u16_t*)alloc((size_t)1024 * 256 * 2);
  float* x_f = (float*)alloc((size_t)1024 * 256 * 4);
  u16_t* x_h = (u16_t*)alloc((size_t)1024 * 256 * 2);
  u16_t* u1 = (u16_t*)alloc((size_t)1024 * 256 * 2);
  float* x2 = (float*)alloc((size_t)1024 * 256 * 4);
  float* gbuf = (float*)alloc((size_t)16 * 256 * 4);
  u16_t* cat2 = (u16_t*)alloc((size_t)1024 * 512 * 2);
  u16_t* v1 = (u16_t*)alloc((size_t)1024 * 256 * 2);
  float* inp = (float*)alloc((size_t)1024 * 256 * 4);
  u16_t* i1 = (u16_t*)alloc((size_t)1024 * 256 * 2);
  u16_t* qkv = (u16_t*)alloc((size_t)1024 * 768 * 2);
  u16_t* vt2 = (u16_t*)alloc((size_t)16 * 256 * 64 * 2);
  float* s2 = (float*)alloc((size_t)16 * 4 * 64 * 64 * 4);
  u16_t* p2 = (u16_t*)alloc((size_t)16 * 4 * 64 * 64 * 2);
  u16_t* sa_o = (u16_t*)alloc((size_t)1024 * 256 * 2);
  float* yb = (float*)alloc((size_t)1024 * 256 * 4);
  u16_t* y2n = (u16_t*)alloc((size_t)1024 * 256 * 2);
  u16_t* f1 = (u16_t*)alloc((size_t)1024 * 256 * 2);
  // aliases (lifetimes disjoint)
  u16_t* t2 = nh2;
  u16_t* KVh = cat_knn;
  u16_t* Vt = nh2;
  u16_t* Pm = kv;

  auto cvt = [&](const float* s, u16_t* d, long n) {
    cvt_kernel<<<(unsigned)((n + 255) / 256), 256, 0, stream>>>(s, d, n);
  };
  auto gemm = [&](const u16_t* A, long lda, long sAb, long sAh,
                  const u16_t* Bw, long ldb, long sBb, long sBh,
                  const float* bias, const float* resid,
                  float* Cf, u16_t* Ch, long ldc, long sCb, long sCh,
                  int M, int N, int Kd, int nb, int Hs, float scale, int relu) {
    if (nb == 1 && (M % 128) == 0 && (N % 128) == 0) {
      dim3 g((unsigned)(M / 128), (unsigned)(N / 128), 1);
      gemm_bf16_big_kernel<<<g, 256, 0, stream>>>(A, lda, Bw, ldb, bias, resid,
                                                  Cf, Ch, ldc, Kd, scale, relu);
    } else {
      dim3 g((unsigned)(M / 64), (unsigned)(N / 64), (unsigned)nb);
      gemm_bf16_kernel<<<g, 256, 0, stream>>>(A, lda, sAb, sAh, Bw, ldb, sBb, sBh,
                                              bias, resid, Cf, Ch, ldc, sCb, sCh,
                                              Kd, Hs, scale, relu);
    }
  };

  // weight conversions
  cvt(d3w2, wb_d3w2, 256 * 64);   cvt(d3w3, wb_d3w3, 256 * 256);
  cvt(d4w2, wb_d4w2, 256 * 64);   cvt(d4w3, wb_d4w3, 256 * 256);
  cvt(d1w1, wb_d1w1, 256 * 512);  cvt(d1w2, wb_d1w2, 256 * 256);
  cvt(d2w1, wb_d2w1, 256 * 512);  cvt(d2w2, wb_d2w2, 256 * 256);
  cvt(ca_win, wb_cawin, 768 * 256); cvt(ca_wout, wb_cawout, 256 * 256);
  cvt(sa_win, wb_sawin, 768 * 256); cvt(sa_wout, wb_sawout, 256 * 256);
  cvt(ffn_w1, wb_ffn1, 256 * 256);  cvt(ffn_w2, wb_ffn2, 256 * 256);
  cvt(fuse_w1, wb_fuse1, 256 * 512); cvt(fuse_w2, wb_fuse2, 256 * 256);
  cvt(fc_w2, wb_fcw2, 256 * 256);
  weff_kernel<<<256, 256, 0, stream>>>(fc_w1, wb_weff);

  // ---- keypoint path: kpt_emb = mlp3(kpt_3d); kpt_comb = mlp2([kpt_feature|kpt_emb]) ----
  mlp_l1_kernel<<<(1024 * 64) / 256, 256, 0, stream>>>(kpt_3d, d3w1, d3b1, kh1, 1024);
  gemm(kh1, 64, 0, 0, wb_d3w2, 64, 0, 0, d3b2, nullptr, nullptr, kh2, 256, 0, 0, 1024, 256, 64, 1, 1, 1.f, 1);
  gemm(kh2, 256, 0, 0, wb_d3w3, 256, 0, 0, d3b3, nullptr, nullptr, cat_kpt + 256, 512, 0, 0, 1024, 256, 256, 1, 1, 1.f, 0);
  cvt_strided_kernel<<<(1024 * 256) / 256, 256, 0, stream>>>(kpt_feature, 256, cat_kpt, 512, 1024, 256);
  gemm(cat_kpt, 512, 0, 0, wb_d1w1, 512, 0, 0, d1b1, nullptr, nullptr, t1, 256, 0, 0, 1024, 256, 512, 1, 1, 1.f, 1);
  gemm(t1, 256, 0, 0, wb_d1w2, 256, 0, 0, d1b2, nullptr, kpt_comb, nullptr, 256, 0, 0, 1024, 256, 256, 1, 1, 1.f, 0);

  // ---- knn path: top-16, gather, knn_emb = mlp3, knn_comb = mlp2 ----
  knn_kernel<<<1024, 256, 0, stream>>>(kpt_3d, pts, idxb);
  gather_kernel<<<16384, 256, 0, stream>>>(idxb, pts, obj_feature, knn_xyz, cat_knn);
  mlp_l1_kernel<<<(16384 * 64) / 256, 256, 0, stream>>>(knn_xyz, d4w1, d4b1, nh1, 16384);
  gemm(nh1, 64, 0, 0, wb_d4w2, 64, 0, 0, d4b2, nullptr, nullptr, nh2, 256, 0, 0, 16384, 256, 64, 1, 1, 1.f, 1);
  gemm(nh2, 256, 0, 0, wb_d4w3, 256, 0, 0, d4b3, nullptr, nullptr, cat_knn + 256, 512, 0, 0, 16384, 256, 256, 1, 1, 1.f, 0);
  gemm(cat_knn, 512, 0, 0, wb_d2w1, 512, 0, 0, d2b1, nullptr, nullptr, t2, 256, 0, 0, 16384, 256, 512, 1, 1, 1.f, 1);
  gemm(t2, 256, 0, 0, wb_d2w2, 256, 0, 0, d2b2, nullptr, nullptr, kv, 256, 0, 0, 16384, 256, 256, 1, 1, 1.f, 0);

  // ---- cross-attention (queries collapsed to 64/batch) ----
  ln_kernel<<<1024, 256, 0, stream>>>(kpt_comb, ln1_g, ln1_b, q1);
  gemm(q1, 256, 0, 0, wb_cawin, 256, 0, 0, ca_bin, nullptr, nullptr, Qh, 256, 0, 0, 1024, 256, 256, 1, 1, 1.f, 0);
  gemm(kv, 256, 0, 0, wb_cawin + 256 * 256, 256, 0, 0, ca_bin + 256, nullptr, nullptr, KVh, 512, 0, 0, 16384, 512, 256, 1, 1, 1.f, 0);
  vtrans_kernel<<<(16 * 1024 * 256) / 256, 256, 0, stream>>>(KVh, Vt);
  // scores: per (b,h): (64 x 1024), K=64, scale 1/sqrt(64)
  gemm(Qh, 256, 64 * 256, 64, KVh, 512, 1024 * 512, 64, nullptr, nullptr,
       Sc, nullptr, 1024, 4L * 64 * 1024, 64 * 1024, 64, 1024, 64, 64, 4, 0.125f, 0);
  softmax_kernel<<<16 * 4 * 64, 256, 0, stream>>>(Sc, Pm, 1024);
  // O = P @ V: per (b,h): (64 x 64), K=1024
  gemm(Pm, 1024, 4L * 64 * 1024, 64 * 1024, Vt, 1024, 256 * 1024, 64 * 1024, nullptr, nullptr,
       nullptr, attn_o, 256, 64 * 256, 64, 64, 64, 1024, 64, 4, 1.f, 0);
  // out proj + residual (q_in row == kpt_comb row)
  gemm(attn_o, 256, 0, 0, wb_cawout, 256, 0, 0, ca_bout, kpt_comb,
       x_f, x_h, 256, 0, 0, 1024, 256, 256, 1, 1, 1.f, 0);

  // ---- fc (collapsed Weff) + mean-fuse ----
  gemm(x_h, 256, 0, 0, wb_weff, 256, 0, 0, fc_b1, nullptr, nullptr, u1, 256, 0, 0, 1024, 256, 256, 1, 1, 1.f, 1);
  gemm(u1, 256, 0, 0, wb_fcw2, 256, 0, 0, fc_b2, nullptr, x2, nullptr, 256, 0, 0, 1024, 256, 256, 1, 1, 1.f, 0);
  meang_kernel<<<16, 256, 0, stream>>>(x2, gbuf);
  cat2_kernel<<<(1024 * 512) / 256, 256, 0, stream>>>(x2, gbuf, cat2);
  gemm(cat2, 512, 0, 0, wb_fuse1, 512, 0, 0, fuse_b1, nullptr, nullptr, v1, 256, 0, 0, 1024, 256, 512, 1, 1, 1.f, 1);
  gemm(v1, 256, 0, 0, wb_fuse2, 256, 0, 0, fuse_b2, nullptr, inp, nullptr, 256, 0, 0, 1024, 256, 256, 1, 1, 1.f, 0);

  // ---- self-attention over 64 tokens ----
  ln_kernel<<<1024, 256, 0, stream>>>(inp, ln2_g, ln2_b, i1);
  gemm(i1, 256, 0, 0, wb_sawin, 256, 0, 0, sa_bin, nullptr, nullptr, qkv, 768, 0, 0, 1024, 768, 256, 1, 1, 1.f, 0);
  vtrans2_kernel<<<(16 * 64 * 256) / 256, 256, 0, stream>>>(qkv, vt2);
  gemm(qkv, 768, 64 * 768, 64, qkv + 256, 768, 64 * 768, 64, nullptr, nullptr,
       s2, nullptr, 64, 4L * 64 * 64, 64 * 64, 64, 64, 64, 64, 4, 0.125f, 0);
  softmax_kernel<<<16 * 4 * 64, 256, 0, stream>>>(s2, p2, 64);
  gemm(p2, 64, 4L * 64 * 64, 64 * 64, vt2, 64, 256 * 64, 64 * 64, nullptr, nullptr,
       nullptr, sa_o, 256, 64 * 256, 64, 64, 64, 64, 64, 4, 1.f, 0);
  gemm(sa_o, 256, 0, 0, wb_sawout, 256, 0, 0, sa_bout, inp,
       yb, nullptr, 256, 0, 0, 1024, 256, 256, 1, 1, 1.f, 0);

  // ---- FFN + final residual -> d_out ----
  ln_kernel<<<1024, 256, 0, stream>>>(yb, ln3_g, ln3_b, y2n);
  gemm(y2n, 256, 0, 0, wb_ffn1, 256, 0, 0, ffn_b1, nullptr, nullptr, f1, 256, 0, 0, 1024, 256, 256, 1, 1, 1.f, 1);
  gemm(f1, 256, 0, 0, wb_ffn2, 256, 0, 0, ffn_b2, yb,
       (float*)d_out, nullptr, 256, 0, 0, 1024, 256, 256, 1, 1, 1.f, 0);
}
